// OffsetGenerator_82059645157449
// MI455X (gfx1250) — matrix operand order repaired and checked by reading, not compiled, not run on hardware
//
#include <hip/hip_runtime.h>
#include <hip/hip_bf16.h>

typedef __attribute__((ext_vector_type(16))) __bf16 v16bf;
typedef __attribute__((ext_vector_type(8)))  float  v8f;

#define STR 132  // padded LDS row stride in bf16 elements (128 data + 4 pad)

// ---- WMMA wrapper: D = A(16x32 bf16) * B(32x16 bf16) + C(16x16 f32) ----
__device__ __forceinline__ v8f wmma_bf16(v16bf a, v16bf b, v8f c) {
  return __builtin_amdgcn_wmma_f32_16x16x32_bf16(
      /*neg_a=*/false, a, /*neg_b=*/false, b,
      /*c_mod=*/(short)0, c, /*reuse_a=*/false, /*reuse_b=*/false);
}

// A-matrix fragment (16x32 bf16), ISA layout:
//   lanes 0-15 = rows M, half-wave selects K sub-block.
//   VGPR j holds K = 8*half + 16*(j>>2) + 2*(j&3), +1
__device__ __forceinline__ v16bf load_a_frag(const __bf16* tile, int row, int half, int kbase) {
  const __bf16* p = tile + row * STR + kbase + 8 * half;
  v16bf f;
#pragma unroll
  for (int j = 0; j < 8; ++j) {
    int k = 16 * (j >> 2) + 2 * (j & 3);
    f[2 * j]     = p[k];
    f[2 * j + 1] = p[k + 1];
  }
  return f;
}

// B-matrix fragment (32x16 bf16) from an N-major (transposed) LDS tile:
//   lane n (0-15) holds column N=n; elements 0-7 K=8*half+0..7, elements 8-15 K=16+8*half+0..7,
//   packed 2 per VGPR -> two runs of 8 contiguous bf16 per lane.
__device__ __forceinline__ v16bf load_b_frag(const __bf16* tileT, int col, int half, int kbase) {
  const __bf16* p = tileT + col * STR + kbase + 8 * half;
  v16bf f;
#pragma unroll
  for (int j = 0; j < 16; ++j) f[j] = p[j < 8 ? j : j + 8];
  return f;
}

__device__ __forceinline__ float gelu_exact(float v) {
  return 0.5f * v * (1.0f + erff(v * 0.70710678118654752f));
}

// x:  [B=512][L=32][H=32][D=128] f32
// W1: [L][H][D=128][DL=128] f32   b1: [L][H][128]
// W2: [L][H][DL=128][D=128] f32   b2: [L][H][128]
// out:[B][L][H][D] f32
// grid = 4096 blocks (1024 (l,h) pairs x 4 B-tiles of 128), block = 256 (8 waves)
__global__ __launch_bounds__(256, 1)
void fused_mlp_gelu_kernel(const float* __restrict__ x,
                           const float* __restrict__ W1,
                           const float* __restrict__ b1,
                           const float* __restrict__ W2,
                           const float* __restrict__ b2,
                           float* __restrict__ out)
{
  __shared__ __bf16 sX [128 * STR];  // X tile,  row-major [b_row][d]
  __shared__ __bf16 sW1[128 * STR];  // W1^T:    [e][d]  (N-major for B frags)
  __shared__ __bf16 sW2[128 * STR];  // W2^T:    [d][e]
  __shared__ __bf16 sH [128 * STR];  // GELU(H) tile, row-major [b_row][e]
  __shared__ float  sB1[128];
  __shared__ float  sB2[128];

  const int blk = blockIdx.x;
  const int lh  = blk >> 2;            // (l*32 + h) in 0..1023
  const int b0  = (blk & 3) * 128;     // batch-tile base row
  const int tid = threadIdx.x;

  const float* w1g = W1 + (size_t)lh * (128 * 128);
  const float* w2g = W2 + (size_t)lh * (128 * 128);

  // ---- stage weights (transposed) and X tile into LDS as bf16 ----
  // 16384 f32 each; 256 threads * float4 = 1024 elems/iter -> 16 iters
#pragma unroll 4
  for (int it = 0; it < 16; ++it) {
    const int idx = it * 1024 + tid * 4;
    const int r = idx >> 7, c = idx & 127;

    const float4 a = *(const float4*)(w1g + idx);     // W1[d=r][e=c..c+3]
    sW1[(c + 0) * STR + r] = (__bf16)a.x;
    sW1[(c + 1) * STR + r] = (__bf16)a.y;
    sW1[(c + 2) * STR + r] = (__bf16)a.z;
    sW1[(c + 3) * STR + r] = (__bf16)a.w;

    const float4 b = *(const float4*)(w2g + idx);     // W2[e=r][d=c..c+3]
    sW2[(c + 0) * STR + r] = (__bf16)b.x;
    sW2[(c + 1) * STR + r] = (__bf16)b.y;
    sW2[(c + 2) * STR + r] = (__bf16)b.z;
    sW2[(c + 3) * STR + r] = (__bf16)b.w;

    const float4 v = *(const float4*)(x + ((size_t)(b0 + r) * 1024 + lh) * 128 + c);
    sX[r * STR + c + 0] = (__bf16)v.x;
    sX[r * STR + c + 1] = (__bf16)v.y;
    sX[r * STR + c + 2] = (__bf16)v.z;
    sX[r * STR + c + 3] = (__bf16)v.w;
  }
  if (tid < 128) {
    sB1[tid] = b1[(size_t)lh * 128 + tid];
    sB2[tid] = b2[(size_t)lh * 128 + tid];
  }
  __syncthreads();

  const int wave = tid >> 5;       // 0..7: 16-row strip per wave
  const int lane = tid & 31;
  const int half = lane >> 4;
  const int lm   = lane & 15;
  const int rowBase = wave * 16;

  // ---- GEMM1: H = GELU(X @ W1 + b1), strip rows [rowBase, rowBase+16) ----
  v16bf xa0 = load_a_frag(sX, rowBase + lm, half, 0);
  v16bf xa1 = load_a_frag(sX, rowBase + lm, half, 32);
  v16bf xa2 = load_a_frag(sX, rowBase + lm, half, 64);
  v16bf xa3 = load_a_frag(sX, rowBase + lm, half, 96);

#pragma unroll
  for (int n = 0; n < 8; ++n) {
    const int col = n * 16 + lm;
    v8f acc = {};
    acc = wmma_bf16(xa0, load_b_frag(sW1, col, half, 0),  acc);
    acc = wmma_bf16(xa1, load_b_frag(sW1, col, half, 32), acc);
    acc = wmma_bf16(xa2, load_b_frag(sW1, col, half, 64), acc);
    acc = wmma_bf16(xa3, load_b_frag(sW1, col, half, 96), acc);
    const float bias = sB1[col];
#pragma unroll
    for (int i = 0; i < 8; ++i) {
      float v = gelu_exact(acc[i] + bias);
      sH[(rowBase + i + 8 * half) * STR + col] = (__bf16)v;  // C/D layout scatter
    }
  }

  // ---- GEMM2: out = H @ W2 + b2 ----
  // This wave only reads the sH rows it just wrote; DS ops are in-order per
  // wave, so no workgroup barrier is required here.
  v16bf ha0 = load_a_frag(sH, rowBase + lm, half, 0);
  v16bf ha1 = load_a_frag(sH, rowBase + lm, half, 32);
  v16bf ha2 = load_a_frag(sH, rowBase + lm, half, 64);
  v16bf ha3 = load_a_frag(sH, rowBase + lm, half, 96);

#pragma unroll
  for (int n = 0; n < 8; ++n) {
    const int col = n * 16 + lm;
    v8f acc = {};
    acc = wmma_bf16(ha0, load_b_frag(sW2, col, half, 0),  acc);
    acc = wmma_bf16(ha1, load_b_frag(sW2, col, half, 32), acc);
    acc = wmma_bf16(ha2, load_b_frag(sW2, col, half, 64), acc);
    acc = wmma_bf16(ha3, load_b_frag(sW2, col, half, 96), acc);
    const float bias = sB2[col];
#pragma unroll
    for (int i = 0; i < 8; ++i) {
      const int brow = b0 + rowBase + i + 8 * half;
      out[((size_t)brow * 1024 + lh) * 128 + col] = acc[i] + bias;
    }
  }
}

extern "C" void kernel_launch(void* const* d_in, const int* in_sizes, int n_in,
                              void* d_out, int out_size, void* d_ws, size_t ws_size,
                              hipStream_t stream) {
  (void)in_sizes; (void)n_in; (void)out_size; (void)d_ws; (void)ws_size;
  const float* x  = (const float*)d_in[0];
  const float* W1 = (const float*)d_in[1];
  const float* b1 = (const float*)d_in[2];
  const float* W2 = (const float*)d_in[3];
  const float* b2 = (const float*)d_in[4];
  float* out = (float*)d_out;

  // 1024 (l,h) pairs x 4 batch tiles of 128 rows
  fused_mlp_gelu_kernel<<<dim3(4096), dim3(256), 0, stream>>>(x, W1, b1, W2, b2, out);
}
